// Block_59287728554309
// MI455X (gfx1250) — matrix operand order repaired and checked by reading, not compiled, not run on hardware
//
#include <hip/hip_runtime.h>

// ---------------------------------------------------------------------------
// CDNA5 feature detection (compile-only environment: guard, don't assume)
// ---------------------------------------------------------------------------
#if defined(__has_builtin)
# if __has_builtin(__builtin_amdgcn_global_load_async_to_lds_b128)
#  define HAVE_ASYNC_LDS 1
# endif
#endif
#ifndef HAVE_ASYNC_LDS
# define HAVE_ASYNC_LDS 0
#endif

#define LAS __attribute__((address_space(3)))

// ---------------------------------------------------------------------------
// Types for CDNA5 WMMA (wave32): 16x16x32 bf16 -> f32 accumulate
// ---------------------------------------------------------------------------
typedef __attribute__((ext_vector_type(16))) __bf16          v16bf;
typedef __attribute__((ext_vector_type(8)))  float           v8f;
typedef __attribute__((ext_vector_type(8)))  unsigned short  us8;
typedef __attribute__((ext_vector_type(4)))  int             v4i;   // async-to-LDS b128 operand

union Frag16 {
  v16bf v;
  us8   h[2];
};

__device__ __forceinline__ unsigned short f2bf(float f) {
  unsigned u = __float_as_uint(f);
  u += 0x7FFFu + ((u >> 16) & 1u);          // round-to-nearest-even
  return (unsigned short)(u >> 16);
}

__device__ __forceinline__ v8f wmma_bf16(v16bf a, v16bf b, v8f c) {
  // emits v_wmma_f32_16x16x32_bf16
  return __builtin_amdgcn_wmma_f32_16x16x32_bf16(
      /*neg_a=*/false, a, /*neg_b=*/false, b,
      /*c_mod=*/(short)0, c, /*reuse_a=*/false, /*reuse_b=*/false);
}

// ---------------------------------------------------------------------------
// fp32 -> bf16 linear conversion (enc_x)
// ---------------------------------------------------------------------------
__global__ __launch_bounds__(256) void cvt_kernel(const float* __restrict__ in,
                                                  unsigned short* __restrict__ out,
                                                  size_t n) {
  size_t i = (size_t)blockIdx.x * blockDim.x + threadIdx.x;
  size_t stride = (size_t)gridDim.x * blockDim.x;
  for (; i < n; i += stride) out[i] = f2bf(in[i]);
}

// ---------------------------------------------------------------------------
// fp32 [K,N] -> bf16 transposed [N,K] (weights), tiled through LDS so both
// the global read and write are coalesced.  Grid: (N/32, K/32), 256 threads.
// ---------------------------------------------------------------------------
__global__ __launch_bounds__(256) void cvtT_kernel(const float* __restrict__ in,
                                                   unsigned short* __restrict__ out,
                                                   int K, int N) {
  __shared__ float tile[32][33];
  const int kb = blockIdx.y * 32, nb = blockIdx.x * 32;
  const int tx = threadIdx.x & 31, ty = threadIdx.x >> 5;   // ty: 0..7
#pragma unroll
  for (int i = 0; i < 32; i += 8) {
    int k = kb + ty + i, n = nb + tx;
    tile[ty + i][tx] = (k < K && n < N) ? in[(size_t)k * N + n] : 0.f;
  }
  __syncthreads();
#pragma unroll
  for (int i = 0; i < 32; i += 8) {
    int n = nb + ty + i, k = kb + tx;
    if (n < N && k < K) out[(size_t)n * K + k] = f2bf(tile[tx][ty + i]);
  }
}

// ---------------------------------------------------------------------------
// LayerNorm: one 256-thread block per row, fp32 in -> bf16 out
// ---------------------------------------------------------------------------
__global__ __launch_bounds__(256) void ln_kernel(const float* __restrict__ x,
                                                 const float* __restrict__ g,
                                                 const float* __restrict__ b,
                                                 unsigned short* __restrict__ out,
                                                 int C) {
  size_t row = blockIdx.x;
  const float* xr = x + row * (size_t)C;
  float s = 0.f, s2 = 0.f;
  for (int i = threadIdx.x; i < C; i += 256) { float v = xr[i]; s += v; s2 += v * v; }
#pragma unroll
  for (int off = 16; off > 0; off >>= 1) {
    s  += __shfl_xor(s,  off, 32);
    s2 += __shfl_xor(s2, off, 32);
  }
  __shared__ float sh[2][8];
  int wid = threadIdx.x >> 5, lane = threadIdx.x & 31;
  if (lane == 0) { sh[0][wid] = s; sh[1][wid] = s2; }
  __syncthreads();
  if (threadIdx.x == 0) {
    float a = 0.f, c2 = 0.f;
    for (int i = 0; i < 8; ++i) { a += sh[0][i]; c2 += sh[1][i]; }
    sh[0][0] = a; sh[1][0] = c2;
  }
  __syncthreads();
  float mean = sh[0][0] / C;
  float var  = sh[1][0] / C - mean * mean;
  float rstd = rsqrtf(var + 1e-5f);
  unsigned short* orow = out + row * (size_t)C;
  for (int i = threadIdx.x; i < C; i += 256)
    orow[i] = f2bf((xr[i] - mean) * rstd * g[i] + b[i]);
}

// ---------------------------------------------------------------------------
// WMMA GEMM: C[M,N] = act(A[M,K] @ W[K,N] + bias) (+ residual)
//   A bf16 row-major [M,K]; WT bf16 PRE-TRANSPOSED [N,K].
//   256 threads = 8 waves.  Block tile 128x128, wave tile 32x64 (8 WMMA/slab),
//   K-slab 32 (= WMMA K).  Double-buffered LDS; next slab staged with
//   GLOBAL_LOAD_ASYNC_TO_LDS_B128 (ASYNCcnt) when available, else a register
//   pipeline.  ACT: 0 none, 1 gelu(tanh), 2 relu.
// ---------------------------------------------------------------------------
template <int ACT, bool RES, bool OUTF, bool OUTB>
__global__ __launch_bounds__(256) void gemm_bf16(const unsigned short* __restrict__ A,
                                                 const unsigned short* __restrict__ WT,
                                                 const float* __restrict__ bias,
                                                 const float* __restrict__ res,
                                                 float* __restrict__ outf,
                                                 unsigned short* __restrict__ outb,
                                                 int M, int N, int K) {
  __shared__ __align__(16) unsigned short As[2][128][40];  // [buf][row][k], +8 pad
  __shared__ __align__(16) unsigned short Bs[2][128][40];  // [buf][n][k],   +8 pad
  const int rowBase = blockIdx.y * 128;
  const int colBase = blockIdx.x * 128;
  const int tid = threadIdx.x;
  const int lane = tid & 31, wid = tid >> 5;
  const int wm = (wid >> 1) * 32, wn = (wid & 1) * 64;
  const int half = lane >> 4, r = lane & 15;
  // staging geometry: each tile = 512 x 16B chunks; 2 chunks/thread/tile
  const int srow = tid >> 2;         // 0..63  (+64 on second chunk)
  const int scol = (tid & 3) * 8;    // 0,8,16,24
  const int nk = K >> 5;
  const us8 zero = {0, 0, 0, 0, 0, 0, 0, 0};
  (void)zero;

  v8f zf = {0.f, 0.f, 0.f, 0.f, 0.f, 0.f, 0.f, 0.f};
  v8f acc[2][4];
#pragma unroll
  for (int am = 0; am < 2; ++am)
#pragma unroll
    for (int j = 0; j < 4; ++j) acc[am][j] = zf;

#if HAVE_ASYNC_LDS
  auto stage = [&](int k0, int buf) {
#pragma unroll
    for (int i = 0; i < 2; ++i) {
      int ar = srow + i * 64;
      int gr = rowBase + ar;
      if (gr < M)
        __builtin_amdgcn_global_load_async_to_lds_b128(
            (v4i*)(A + (size_t)gr * K + k0 + scol),
            (LAS v4i*)&As[buf][ar][scol], 0, 0);
      __builtin_amdgcn_global_load_async_to_lds_b128(
          (v4i*)(WT + (size_t)(colBase + ar) * K + k0 + scol),
          (LAS v4i*)&Bs[buf][ar][scol], 0, 0);
    }
  };
#else
  us8 aR[2], bR[2];
  auto loadRegs = [&](int k0) {
#pragma unroll
    for (int i = 0; i < 2; ++i) {
      int ar = srow + i * 64;
      int gr = rowBase + ar;
      aR[i] = (gr < M) ? *(const us8*)(A + (size_t)gr * K + k0 + scol) : zero;
      bR[i] = *(const us8*)(WT + (size_t)(colBase + ar) * K + k0 + scol);
    }
  };
  auto storeRegs = [&](int buf) {
#pragma unroll
    for (int i = 0; i < 2; ++i) {
      int ar = srow + i * 64;
      *(us8*)&As[buf][ar][scol] = aR[i];
      *(us8*)&Bs[buf][ar][scol] = bR[i];
    }
  };
#endif

  auto compute = [&](int cur) {
    Frag16 af[2], bfr[4];
    const unsigned short* p;
    // A-frag per ISA 16-bit layout: lanes 0-15 -> K{0..7,16..23}, lanes 16-31 -> K{8..15,24..31}
    p = &As[cur][wm + r][0];
    af[0].h[0] = *(const us8*)(p + half * 8); af[0].h[1] = *(const us8*)(p + 16 + half * 8);
    p = &As[cur][wm + 16 + r][0];
    af[1].h[0] = *(const us8*)(p + half * 8); af[1].h[1] = *(const us8*)(p + 16 + half * 8);
    // B-frag: lane%16 = N, lanes 0-15 hold K{0..7,16..23}, lanes 16-31 hold K{8..15,24..31}
#pragma unroll
    for (int j = 0; j < 4; ++j) {
      p = &Bs[cur][wn + j * 16 + r][0];
      bfr[j].h[0] = *(const us8*)(p + half * 8);
      bfr[j].h[1] = *(const us8*)(p + 16 + half * 8);
    }
#pragma unroll
    for (int am = 0; am < 2; ++am)
#pragma unroll
      for (int j = 0; j < 4; ++j)
        acc[am][j] = wmma_bf16(af[am].v, bfr[j].v, acc[am][j]);
  };

#if HAVE_ASYNC_LDS
  stage(0, 0);
  asm volatile("s_wait_asynccnt 0x0" ::: "memory");
  __syncthreads();
  for (int k = 0; k < nk; ++k) {
    if (k + 1 < nk) stage((k + 1) * 32, (k + 1) & 1);   // overlap DMA with WMMA
    compute(k & 1);
    asm volatile("s_wait_asynccnt 0x0" ::: "memory");
    __syncthreads();
  }
#else
  loadRegs(0);
  storeRegs(0);
  for (int k = 0; k < nk; ++k) {
    if (k + 1 < nk) loadRegs((k + 1) * 32);
    __syncthreads();
    compute(k & 1);
    if (k + 1 < nk) storeRegs((k + 1) & 1);
  }
#endif

  // epilogue: bias -> activation -> residual -> store
#pragma unroll
  for (int am = 0; am < 2; ++am) {
#pragma unroll
    for (int j = 0; j < 4; ++j) {
      int n = colBase + wn + j * 16 + r;
      float bv = bias[n];
      v8f a = acc[am][j];
#pragma unroll
      for (int e = 0; e < 8; ++e) {
        int m = rowBase + wm + am * 16 + e + half * 8;  // C-layout: lane>=16 -> M+8
        if (m < M) {
          float v = a[e] + bv;
          if (ACT == 1) { float t = v; v = 0.5f * t * (1.f + tanhf(0.7978845608f * (t + 0.044715f * t * t * t))); }
          if (ACT == 2) { v = fmaxf(v, 0.f); }
          if (RES) v += res[(size_t)m * N + n];
          size_t idx = (size_t)m * N + n;
          if (OUTF) outf[idx] = v;
          if (OUTB) outb[idx] = f2bf(v);
        }
      }
    }
  }
}

// ---------------------------------------------------------------------------
// Flash attention (bf16 in, bf16 out), head dim D=64, WMMA for S and P*V.
//   Block: 128 threads = 4 waves, 64 Q rows per block (16 per wave).
//   KV tiles of 32 staged in LDS (K via async-to-LDS when available);
//   P transposed through per-wave LDS with an explicit s_wait_dscnt.
// ---------------------------------------------------------------------------
template <bool CAUSAL>
__global__ __launch_bounds__(128) void flash_attn(const unsigned short* __restrict__ Q,
                                                  const unsigned short* __restrict__ Kp,
                                                  const unsigned short* __restrict__ Vp,
                                                  unsigned short* __restrict__ O,
                                                  int Ts, int Hh,
                                                  int qStride, int kvStride, int oStride,
                                                  size_t qBatch, size_t kvBatch, size_t oBatch,
                                                  float scale) {
  __shared__ __align__(16) unsigned short Ks[32][72];     // [s][d]  (natural B-layout for Q*K^T)
  __shared__ __align__(16) unsigned short Vs[64][40];     // [d][s]  (B-layout for P*V)
  __shared__ __align__(16) unsigned short Ps[4][16][40];  // per-wave P transpose buffer
  const int b = blockIdx.x / Hh, h = blockIdx.x % Hh;
  const int lane = threadIdx.x & 31, wid = threadIdx.x >> 5;
  const int half = lane >> 4, r = lane & 15;
  const int qrow0 = blockIdx.y * 64 + wid * 16;
  const us8 zero = {0, 0, 0, 0, 0, 0, 0, 0};
  v8f zf = {0.f, 0.f, 0.f, 0.f, 0.f, 0.f, 0.f, 0.f};

  const unsigned short* Qb = Q + (size_t)b * qBatch + h * 64;
  const unsigned short* Kb = Kp + (size_t)b * kvBatch + h * 64;
  const unsigned short* Vb = Vp + (size_t)b * kvBatch + h * 64;

  // Q fragments (16 rows x 64 d = two k-steps), loaded once
  Frag16 aQ[2];
#pragma unroll
  for (int aq = 0; aq < 2; ++aq) {
    const unsigned short* qp = Qb + (size_t)(qrow0 + r) * qStride + aq * 32;
    aQ[aq].h[0] = *(const us8*)(qp + half * 8);
    aQ[aq].h[1] = *(const us8*)(qp + 16 + half * 8);
  }

  v8f o0 = zf, o1 = zf, o2 = zf, o3 = zf;   // O accum 16x64
  float m[8], l[8];
#pragma unroll
  for (int e = 0; e < 8; ++e) { m[e] = -1e30f; l[e] = 0.f; }

  int nT = (Ts + 31) >> 5;
  if (CAUSAL) {
    int lim = ((int)blockIdx.y * 64 + 63) / 32 + 1;  // skip fully-masked tiles
    if (lim < nT) nT = lim;
  }

  for (int t = 0; t < nT; ++t) {
    int st = t * 32;
    // cooperative KV stage: K natural (async path), V transposed manually
#pragma unroll
    for (int i = 0; i < 2; ++i) {
      int c = threadIdx.x + i * 128;
      int sr = c >> 3;
      int dc = (c & 7) * 8;
      int gs = st + sr;
      us8 vv = zero;
      if (gs < Ts) {
#if HAVE_ASYNC_LDS
        __builtin_amdgcn_global_load_async_to_lds_b128(
            (v4i*)(Kb + (size_t)gs * kvStride + dc),
            (LAS v4i*)&Ks[sr][dc], 0, 0);
#else
        *(us8*)&Ks[sr][dc] = *(const us8*)(Kb + (size_t)gs * kvStride + dc);
#endif
        vv = *(const us8*)(Vb + (size_t)gs * kvStride + dc);
      }
#if !HAVE_ASYNC_LDS
      else {
        *(us8*)&Ks[sr][dc] = zero;
      }
#endif
#pragma unroll
      for (int j = 0; j < 8; ++j) Vs[dc + j][sr] = vv[j];
    }
#if HAVE_ASYNC_LDS
    asm volatile("s_wait_asynccnt 0x0" ::: "memory");
#endif
    __syncthreads();

    // S = Q * K^T  (16x32 per wave)
    v8f s0 = zf, s1 = zf;
#pragma unroll
    for (int ks = 0; ks < 2; ++ks) {
      Frag16 bK0, bK1;
      const unsigned short* kp0 = &Ks[r][ks * 32 + half * 8];
      bK0.h[0] = *(const us8*)kp0; bK0.h[1] = *(const us8*)(kp0 + 16);
      const unsigned short* kp1 = &Ks[16 + r][ks * 32 + half * 8];
      bK1.h[0] = *(const us8*)kp1; bK1.h[1] = *(const us8*)(kp1 + 16);
      s0 = wmma_bf16(aQ[ks].v, bK0.v, s0);
      s1 = wmma_bf16(aQ[ks].v, bK1.v, s1);
    }

    // online softmax update + P write to per-wave LDS (C-layout -> A-layout transpose)
#pragma unroll
    for (int e = 0; e < 8; ++e) {
      int rowq = qrow0 + e + half * 8;
      int c0 = st + r, c1 = st + 16 + r;
      float v0 = s0[e] * scale, v1 = s1[e] * scale;
      if (c0 >= Ts || (CAUSAL && c0 > rowq)) v0 = -1e30f;
      if (c1 >= Ts || (CAUSAL && c1 > rowq)) v1 = -1e30f;
      float mx = fmaxf(v0, v1);
#pragma unroll
      for (int off = 1; off < 16; off <<= 1) mx = fmaxf(mx, __shfl_xor(mx, off, 32));
      float mn = fmaxf(m[e], mx);
      float f = __expf(m[e] - mn);
      m[e] = mn;
      float p0 = __expf(v0 - mn), p1 = __expf(v1 - mn);
      float ps = p0 + p1;
#pragma unroll
      for (int off = 1; off < 16; off <<= 1) ps += __shfl_xor(ps, off, 32);
      l[e] = l[e] * f + ps;
      o0[e] *= f; o1[e] *= f; o2[e] *= f; o3[e] *= f;
      Ps[wid][e + half * 8][r]      = f2bf(p0);
      Ps[wid][e + half * 8][16 + r] = f2bf(p1);
    }
    // DS ops are in-order per wave; wait for the transpose stores before re-reading
    asm volatile("s_wait_dscnt 0x0" ::: "memory");

    Frag16 aP;
    const unsigned short* pp = &Ps[wid][r][0];
    aP.h[0] = *(const us8*)(pp + half * 8);
    aP.h[1] = *(const us8*)(pp + 16 + half * 8);

    // O += P * V  (16x64, K=32)
    {
      Frag16 bV;
      const unsigned short* vp;
      vp = &Vs[r][half * 8];      bV.h[0] = *(const us8*)vp; bV.h[1] = *(const us8*)(vp + 16);
      o0 = wmma_bf16(aP.v, bV.v, o0);
      vp = &Vs[16 + r][half * 8]; bV.h[0] = *(const us8*)vp; bV.h[1] = *(const us8*)(vp + 16);
      o1 = wmma_bf16(aP.v, bV.v, o1);
      vp = &Vs[32 + r][half * 8]; bV.h[0] = *(const us8*)vp; bV.h[1] = *(const us8*)(vp + 16);
      o2 = wmma_bf16(aP.v, bV.v, o2);
      vp = &Vs[48 + r][half * 8]; bV.h[0] = *(const us8*)vp; bV.h[1] = *(const us8*)(vp + 16);
      o3 = wmma_bf16(aP.v, bV.v, o3);
    }
    __syncthreads();
  }

  unsigned short* Ob = O + (size_t)b * oBatch;
#pragma unroll
  for (int e = 0; e < 8; ++e) {
    float inv = 1.0f / l[e];
    size_t rowo = (size_t)(qrow0 + e + half * 8) * oStride + h * 64;
    Ob[rowo + r]      = f2bf(o0[e] * inv);
    Ob[rowo + 16 + r] = f2bf(o1[e] * inv);
    Ob[rowo + 32 + r] = f2bf(o2[e] * inv);
    Ob[rowo + 48 + r] = f2bf(o3[e] * inv);
  }
}

// ---------------------------------------------------------------------------
// Launch: full decoder block pipeline
// ---------------------------------------------------------------------------
extern "C" void kernel_launch(void* const* d_in, const int* in_sizes, int n_in,
                              void* d_out, int out_size, void* d_ws, size_t ws_size,
                              hipStream_t stream) {
  (void)in_sizes; (void)n_in; (void)out_size; (void)ws_size;
  constexpr int Bc = 8, T = 1024, T2 = 257, C = 1024, Hh = 16, HID = 256;
  constexpr int M  = Bc * T;    // 8192
  constexpr int M2 = Bc * T2;   // 2056

  const float* x      = (const float*)d_in[0];
  const float* enc_x  = (const float*)d_in[1];
  const float* ln1_g  = (const float*)d_in[2];
  const float* ln1_b  = (const float*)d_in[3];
  const float* ln2_g  = (const float*)d_in[4];
  const float* ln2_b  = (const float*)d_in[5];
  const float* ln3_g  = (const float*)d_in[6];
  const float* ln3_b  = (const float*)d_in[7];
  const float* qkv_w  = (const float*)d_in[8];
  const float* qkv_b  = (const float*)d_in[9];
  const float* proj_w = (const float*)d_in[10];
  const float* proj_b = (const float*)d_in[11];
  const float* caq_w  = (const float*)d_in[12];
  const float* caq_b  = (const float*)d_in[13];
  const float* cak_w  = (const float*)d_in[14];
  const float* cak_b  = (const float*)d_in[15];
  const float* cav_w  = (const float*)d_in[16];
  const float* cav_b  = (const float*)d_in[17];
  const float* cap_w  = (const float*)d_in[18];
  const float* cap_b  = (const float*)d_in[19];
  const float* fc_w   = (const float*)d_in[20];
  const float* fc_b   = (const float*)d_in[21];
  const float* mlp_w  = (const float*)d_in[22];
  const float* mlp_b  = (const float*)d_in[23];
  const float* add_w  = (const float*)d_in[24];
  const float* add_b  = (const float*)d_in[25];
  const float* adu_w  = (const float*)d_in[26];
  const float* adu_b  = (const float*)d_in[27];

  char* ws = (char*)d_ws;
  size_t off = 0;
  auto alloc = [&](size_t elems, size_t esz) -> char* {
    char* p = ws + off;
    off += (elems * esz + 255) & ~(size_t)255;
    return p;
  };
  // weights stored bf16 TRANSPOSED: [N][K]
  unsigned short* wQkv = (unsigned short*)alloc((size_t)C * 3 * C, 2);
  unsigned short* wPrj = (unsigned short*)alloc((size_t)C * C, 2);
  unsigned short* wCaQ = (unsigned short*)alloc((size_t)C * C, 2);
  unsigned short* wCaK = (unsigned short*)alloc((size_t)C * C, 2);
  unsigned short* wCaV = (unsigned short*)alloc((size_t)C * C, 2);
  unsigned short* wCaP = (unsigned short*)alloc((size_t)C * C, 2);
  unsigned short* wFc  = (unsigned short*)alloc((size_t)C * 4 * C, 2);
  unsigned short* wMlp = (unsigned short*)alloc((size_t)4 * C * C, 2);
  unsigned short* wAdD = (unsigned short*)alloc((size_t)C * HID, 2);
  unsigned short* wAdU = (unsigned short*)alloc((size_t)HID * C, 2);
  unsigned short* encB = (unsigned short*)alloc((size_t)M2 * C, 2);
  unsigned short* hB   = (unsigned short*)alloc((size_t)M * C, 2);
  unsigned short* qkvB = (unsigned short*)alloc((size_t)M * 3 * C, 2);
  unsigned short* OB   = (unsigned short*)alloc((size_t)M * C, 2);
  float*          xacc = (float*)alloc((size_t)M * C, 4);
  unsigned short* ffnB = (unsigned short*)alloc((size_t)M * 4 * C, 2);
  unsigned short* adB  = (unsigned short*)alloc((size_t)M * HID, 2);

  auto cvtT = [&](const float* s, unsigned short* d, int K, int N) {
    cvtT_kernel<<<dim3(N / 32, K / 32), 256, 0, stream>>>(s, d, K, N);
  };
  cvtT(qkv_w, wQkv, C, 3 * C);
  cvtT(proj_w, wPrj, C, C);
  cvtT(caq_w, wCaQ, C, C);
  cvtT(cak_w, wCaK, C, C);
  cvtT(cav_w, wCaV, C, C);
  cvtT(cap_w, wCaP, C, C);
  cvtT(fc_w, wFc, C, 4 * C);
  cvtT(mlp_w, wMlp, 4 * C, C);
  cvtT(add_w, wAdD, C, HID);
  cvtT(adu_w, wAdU, HID, C);
  cvt_kernel<<<2048, 256, 0, stream>>>(enc_x, encB, (size_t)M2 * C);

  // --- causal self-attention ---
  ln_kernel<<<M, 256, 0, stream>>>(x, ln1_g, ln1_b, hB, C);
  gemm_bf16<0, false, false, true><<<dim3(3 * C / 128, M / 128), 256, 0, stream>>>(
      hB, wQkv, qkv_b, nullptr, nullptr, qkvB, M, 3 * C, C);
  flash_attn<true><<<dim3(Bc * Hh, T / 64), 128, 0, stream>>>(
      qkvB, qkvB + C, qkvB + 2 * C, OB, T, Hh, 3 * C, 3 * C, C,
      (size_t)T * 3 * C, (size_t)T * 3 * C, (size_t)T * C, 0.125f);
  gemm_bf16<0, true, true, false><<<dim3(C / 128, M / 128), 256, 0, stream>>>(
      OB, wPrj, proj_b, x, xacc, nullptr, M, C, C);

  // --- cross-attention ---
  ln_kernel<<<M, 256, 0, stream>>>(xacc, ln3_g, ln3_b, hB, C);
  unsigned short* q2B = qkvB;
  unsigned short* k2B = qkvB + (size_t)M * C;
  unsigned short* v2B = k2B + (size_t)M2 * C;
  gemm_bf16<0, false, false, true><<<dim3(C / 128, M / 128), 256, 0, stream>>>(
      hB, wCaQ, caq_b, nullptr, nullptr, q2B, M, C, C);
  gemm_bf16<0, false, false, true><<<dim3(C / 128, (M2 + 127) / 128), 256, 0, stream>>>(
      encB, wCaK, cak_b, nullptr, nullptr, k2B, M2, C, C);
  gemm_bf16<0, false, false, true><<<dim3(C / 128, (M2 + 127) / 128), 256, 0, stream>>>(
      encB, wCaV, cav_b, nullptr, nullptr, v2B, M2, C, C);
  flash_attn<false><<<dim3(Bc * Hh, T / 64), 128, 0, stream>>>(
      q2B, k2B, v2B, OB, T2, Hh, C, C, C,
      (size_t)T * C, (size_t)T2 * C, (size_t)T * C, 0.125f);
  gemm_bf16<0, true, true, false><<<dim3(C / 128, M / 128), 256, 0, stream>>>(
      OB, wCaP, cap_b, xacc, xacc, nullptr, M, C, C);

  // --- MLP ---
  ln_kernel<<<M, 256, 0, stream>>>(xacc, ln2_g, ln2_b, hB, C);
  gemm_bf16<1, false, false, true><<<dim3(4 * C / 128, M / 128), 256, 0, stream>>>(
      hB, wFc, fc_b, nullptr, nullptr, ffnB, M, 4 * C, C);
  gemm_bf16<0, true, true, true><<<dim3(C / 128, M / 128), 256, 0, stream>>>(
      ffnB, wMlp, mlp_b, xacc, xacc, hB, M, C, 4 * C);  // hB = bf16(x3) for adapter

  // --- adapter ---
  gemm_bf16<2, false, false, true><<<dim3(HID / 128, M / 128), 256, 0, stream>>>(
      hB, wAdD, add_b, nullptr, nullptr, adB, M, HID, C);
  gemm_bf16<0, true, true, false><<<dim3(C / 128, M / 128), 256, 0, stream>>>(
      adB, wAdU, adu_b, xacc, (float*)d_out, nullptr, M, C, HID);

  // second tuple output: enc_x passthrough
  (void)hipMemcpyAsync((float*)d_out + (size_t)M * C, enc_x,
                       (size_t)M2 * C * sizeof(float), hipMemcpyDeviceToDevice, stream);
}
